// Linear_7206955122813
// MI455X (gfx1250) — 8-bit matrix operand order repaired and checked by reading, compiled, not run on hardware
//
#include <hip/hip_runtime.h>
#include <hip/hip_fp16.h>

typedef int    v16i __attribute__((ext_vector_type(16)));
typedef float  v8f  __attribute__((ext_vector_type(8)));

#define FP8E5M2_MAX 3.5f

__device__ __forceinline__ unsigned int pack4_e5m2(float a, float b, float c, float d, float inv) {
    float f[4] = {a, b, c, d};
    unsigned int r = 0;
#pragma unroll
    for (int i = 0; i < 4; ++i) {
        __half h = __float2half(f[i] * inv);
        unsigned short u = __half_as_ushort(h);
        unsigned short ru = (unsigned short)(u + 0x7F + ((u >> 8) & 1));
        r |= ((unsigned int)(ru >> 8) & 0xFFu) << (8 * i);
    }
    return r;
}

__global__ __launch_bounds__(256)
void quant_rows_e5m2(const float* __restrict__ x, unsigned char* __restrict__ q,
                     float* __restrict__ s, int K) {
    const int row = blockIdx.x;
    const float* xr = x + (size_t)row * K;

    float amax = 0.0f;
    for (int k = threadIdx.x * 4; k < K; k += blockDim.x * 4) {
        float4 v = *(const float4*)(xr + k);
        amax = fmaxf(amax, fmaxf(fmaxf(fabsf(v.x), fabsf(v.y)),
                                 fmaxf(fabsf(v.z), fabsf(v.w))));
    }
#pragma unroll
    for (int off = 16; off > 0; off >>= 1)
        amax = fmaxf(amax, __shfl_xor(amax, off, 32));

    __shared__ float smax[8];
    const int wid  = threadIdx.x >> 5;
    const int lane = threadIdx.x & 31;
    if (lane == 0) smax[wid] = amax;
    __syncthreads();
    if (wid == 0) {
        float m = (lane < 8) ? smax[lane] : 0.0f;
#pragma unroll
        for (int off = 4; off > 0; off >>= 1)
            m = fmaxf(m, __shfl_xor(m, off, 32));
        if (lane == 0) smax[0] = m;
    }
    __syncthreads();

    const float scale = fmaxf(smax[0], 1e-12f) / FP8E5M2_MAX;
    const float inv   = 1.0f / scale;
    if (threadIdx.x == 0) s[row] = scale;

    unsigned int* qo = (unsigned int*)(q + (size_t)row * K);
    for (int k = threadIdx.x * 4; k < K; k += blockDim.x * 4) {
        float4 v = *(const float4*)(xr + k);
        qo[k >> 2] = pack4_e5m2(v.x, v.y, v.z, v.w, inv);
    }
}

#if __has_builtin(__builtin_amdgcn_wmma_f32_16x16x128_bf8_bf8)
#define WMMA_E5M2(A, B, C) \
    __builtin_amdgcn_wmma_f32_16x16x128_bf8_bf8((A), (B), (short)0, (C), false, false)
#else
#define WMMA_E5M2(A, B, C) \
    __builtin_amdgcn_wmma_f32_16x16x128_fp8_fp8((A), (B), (short)0, (C), false, false)
#endif

__device__ __forceinline__ void load_afrags(v16i (&af)[2], const unsigned char* ap, int K) {
#pragma unroll
    for (int mt = 0; mt < 2; ++mt) {
#pragma unroll
        for (int p = 0; p < 8; ++p) {
            uint2 d = *(const uint2*)(ap + (size_t)mt * 16 * K
                                         + (p >> 1) * 32 + (p & 1) * 8);
            af[mt][2 * p]     = (int)d.x;
            af[mt][2 * p + 1] = (int)d.y;
        }
    }
}

__device__ __forceinline__ void load_bfrags(v16i (&bf)[4], const unsigned char* bp, int K) {
#pragma unroll
    for (int nt = 0; nt < 4; ++nt) {
#pragma unroll
        for (int qd = 0; qd < 4; ++qd) {
            uint4 d = *(const uint4*)(bp + (size_t)nt * 16 * K + qd * 32);
            bf[nt][4 * qd]     = (int)d.x;
            bf[nt][4 * qd + 1] = (int)d.y;
            bf[nt][4 * qd + 2] = (int)d.z;
            bf[nt][4 * qd + 3] = (int)d.w;
        }
    }
}

__device__ __forceinline__ void compute_chunk(v8f (&acc)[2][4], v16i (&af)[2], v16i (&bf)[4]) {
#pragma unroll
    for (int nt = 0; nt < 4; ++nt) {
        acc[0][nt] = WMMA_E5M2(af[0], bf[nt], acc[0][nt]);
        acc[1][nt] = WMMA_E5M2(af[1], bf[nt], acc[1][nt]);
    }
}

template <int K_CONST>
__global__ __launch_bounds__(256)
void gemm_bf8_scaled(const unsigned char* __restrict__ qa,
                     const unsigned char* __restrict__ qw,
                     const float* __restrict__ sa,
                     const float* __restrict__ sw,
                     const float* __restrict__ bias,
                     float* __restrict__ out,
                     int M, int N, int Kdyn) {
    const int K = (K_CONST > 0) ? K_CONST : Kdyn;

    const int lane  = threadIdx.x & 31;
    const int wave  = threadIdx.x >> 5;
    const int half  = lane >> 4;
    const int l16   = lane & 15;

    const int waveM = wave >> 1;
    const int waveN = wave & 1;
    const int m0 = blockIdx.y * 128 + waveM * 32;
    const int n0 = blockIdx.x * 128 + waveN * 64;

    v8f acc[2][4];
#pragma unroll
    for (int mt = 0; mt < 2; ++mt)
#pragma unroll
        for (int nt = 0; nt < 4; ++nt)
            acc[mt][nt] = (v8f){0.f, 0.f, 0.f, 0.f, 0.f, 0.f, 0.f, 0.f};

    const unsigned char* ap = qa + (size_t)(m0 + l16) * K + half * 16;
    const unsigned char* bp = qw + (size_t)(n0 + l16) * K + half * 16;

    if constexpr (K_CONST > 0) {
        constexpr int kChunks = K_CONST >> 7;
        static_assert((kChunks & 1) == 0 && kChunks >= 2, "K must be multiple of 256");

        v16i a0[2], b0[4], a1[2], b1[4];
        load_afrags(a0, ap, K);
        load_bfrags(b0, bp, K);

        for (int kc = 0; kc < kChunks - 2; kc += 2) {
            ap += 128; bp += 128;
            __builtin_prefetch(ap + 128, 0, 1);
            __builtin_prefetch(bp + 128, 0, 1);
            load_afrags(a1, ap, K);
            load_bfrags(b1, bp, K);
            compute_chunk(acc, a0, b0);

            ap += 128; bp += 128;
            __builtin_prefetch(ap + 128, 0, 1);
            __builtin_prefetch(bp + 128, 0, 1);
            load_afrags(a0, ap, K);
            load_bfrags(b0, bp, K);
            compute_chunk(acc, a1, b1);
        }
        ap += 128; bp += 128;
        load_afrags(a1, ap, K);
        load_bfrags(b1, bp, K);
        compute_chunk(acc, a0, b0);
        compute_chunk(acc, a1, b1);
    } else {
        const int kChunks = K >> 7;
        for (int kc = 0; kc < kChunks; ++kc) {
            v16i af[2], bf[4];
            load_afrags(af, ap, K);
            load_bfrags(bf, bp, K);
            compute_chunk(acc, af, bf);
            ap += 128; bp += 128;
        }
    }

#pragma unroll
    for (int mt = 0; mt < 2; ++mt) {
#pragma unroll
        for (int nt = 0; nt < 4; ++nt) {
            const int n     = n0 + nt * 16 + l16;
            const float swn = sw[n];
            const float bn  = bias[n];
#pragma unroll
            for (int r = 0; r < 8; ++r) {
                const int m = m0 + mt * 16 + half * 8 + r;
                out[(size_t)m * N + n] = acc[mt][nt][r] * sa[m] * swn + bn;
            }
        }
    }
}

extern "C" void kernel_launch(void* const* d_in, const int* in_sizes, int n_in,
                              void* d_out, int out_size, void* d_ws, size_t ws_size,
                              hipStream_t stream) {
    const float* x    = (const float*)d_in[0];
    const float* wgt  = (const float*)d_in[1];
    const float* bias = (const float*)d_in[2];
    float* out = (float*)d_out;

    const int N = in_sizes[2];
    const int K = in_sizes[1] / N;
    const int M = in_sizes[0] / K;

    unsigned char* qa = (unsigned char*)d_ws;
    unsigned char* qw = qa + (size_t)M * K;
    float* sa = (float*)(qw + (size_t)N * K);
    float* sw = sa + M;

    quant_rows_e5m2<<<M, 256, 0, stream>>>(x,   qa, sa, K);
    quant_rows_e5m2<<<N, 256, 0, stream>>>(wgt, qw, sw, K);

    dim3 grid(N / 128, M / 128);
    if (K == 4096) {
        gemm_bf8_scaled<4096><<<grid, 256, 0, stream>>>(qa, qw, sa, sw, bias, out, M, N, K);
    } else {
        gemm_bf8_scaled<0><<<grid, 256, 0, stream>>>(qa, qw, sa, sw, bias, out, M, N, K);
    }
}
